// RegionProposalNetwork_61160334295670
// MI455X (gfx1250) — compile-verified, matrix operand order checked by reading, not run on hardware
//
#include <hip/hip_runtime.h>
#include <hip/hip_bf16.h>
#include <stdint.h>

// ---------------- types ----------------
typedef __attribute__((ext_vector_type(4)))  unsigned int  u32x4;
typedef __attribute__((ext_vector_type(8)))  int           i32x8;
typedef __attribute__((ext_vector_type(4)))  int           i32x4;
typedef __attribute__((ext_vector_type(8)))  float         f32x8;
typedef __attribute__((ext_vector_type(8)))  __bf16        bf16x8;
typedef __attribute__((ext_vector_type(16))) __bf16        bf16x16;

// ---------------- problem constants ----------------
#define B_   4
#define C_   256
#define H_   200
#define W_   336
#define CO_  512
#define TILE_M 16
#define WTILES (W_ / TILE_M)            // 21 (336 = 21*16, exact)
#define HEADN 64                        // 18 cls + 36 box, padded to 64
#define CLS_TOTAL ((size_t)B_*H_*W_*18)

// workspace layout (bytes)
#define X2_BYTES  ((unsigned long long)B_*H_*W_*C_*2ull)     // NHWC bf16 input
#define WT2_OFF   ((X2_BYTES + 255ull) & ~255ull)
#define WT2_BYTES (9ull*CO_*C_*2ull)                          // [kk][o][c] bf16
#define HW2_OFF   ((WT2_OFF + WT2_BYTES + 255ull) & ~255ull)
#define HW2_BYTES ((unsigned long long)HEADN*CO_*2ull)        // [o][c] bf16

// LDS carve (ushort units)
#define FSTRIDE 528
#define SA_OFF  0                        // 18*32      = 576 ushorts
#define SF_OFF  576                      // 16*528     = 8448 ushorts
#define SHW_OFF 9024                     // 64*512     = 32768 ushorts
#define SMEM_USHORTS (SHW_OFF + HEADN*CO_)

__device__ __forceinline__ unsigned short f32_to_bf16(float f) {
  unsigned u = __float_as_uint(f);
  unsigned r = (u + 0x7FFFu + ((u >> 16) & 1u)) >> 16;
  return (unsigned short)r;
}

__device__ __forceinline__ bf16x16 frag16(const unsigned short* p, int kb) {
  bf16x8 lo = *(const bf16x8*)(p + kb);
  bf16x8 hi = *(const bf16x8*)(p + kb + 16);
  return __builtin_shufflevector(lo, hi, 0,1,2,3,4,5,6,7,8,9,10,11,12,13,14,15);
}

// ---------------- kernel 1: NCHW f32 -> NHWC bf16 ----------------
__global__ __launch_bounds__(256) void k_nchw_to_nhwc(const float* __restrict__ x,
                                                      unsigned short* __restrict__ x2) {
  __shared__ float tile[32][33];
  const int b  = blockIdx.z >> 3;
  const int cg = blockIdx.z & 7;
  const int h  = blockIdx.y;
  const int w0 = blockIdx.x * 32;
  const int r  = threadIdx.x >> 5;
  const int l  = threadIdx.x & 31;
#pragma unroll
  for (int i = 0; i < 4; ++i) {
    const int cr = r + i * 8;
    const int c  = cg * 32 + cr;
    const int w  = w0 + l;
    float v = 0.f;
    if (w < W_) v = x[(((size_t)b * C_ + c) * H_ + h) * W_ + w];
    tile[cr][l] = v;
  }
  __syncthreads();
#pragma unroll
  for (int i = 0; i < 4; ++i) {
    const int wr = r + i * 8;
    const int w  = w0 + wr;
    if (w < W_) {
      const int c = cg * 32 + l;
      x2[(((size_t)b * H_ + h) * W_ + w) * C_ + c] = f32_to_bf16(tile[l][wr]);
    }
  }
}

// ---------------- kernel 2: W1 OIHW f32 -> [kk][o][c] bf16 ----------------
__global__ __launch_bounds__(256) void k_w1_repack(const float* __restrict__ W1,
                                                   unsigned short* __restrict__ wt2) {
  const int i  = blockIdx.x * 256 + threadIdx.x;     // 0 .. 9*512*256-1
  const int c  = i & 255;
  const int o  = (i >> 8) & 511;
  const int kk = i >> 17;                            // ky*3+kx
  wt2[i] = f32_to_bf16(W1[((size_t)o * C_ + c) * 9 + kk]);
}

// ---------------- kernel 3: Wc/Wb -> padded [64][512] bf16 ----------------
__global__ __launch_bounds__(256) void k_head_repack(const float* __restrict__ Wc,
                                                     const float* __restrict__ Wb,
                                                     unsigned short* __restrict__ hw2) {
  const int i = blockIdx.x * 256 + threadIdx.x;      // 0..32767
  const int c = i & 511, o = i >> 9;
  float v = 0.f;
  if (o < 18)       v = Wc[o * 512 + c];
  else if (o < 54)  v = Wb[(o - 18) * 512 + c];
  hw2[i] = f32_to_bf16(v);
}

// ---------------- kernel 4: fused conv3x3+ReLU + 1x1 heads ----------------
__global__ __launch_bounds__(256) void rpn_main(const unsigned short* __restrict__ x2,
                                                const unsigned short* __restrict__ wt2,
                                                const unsigned short* __restrict__ hw2,
                                                const float* __restrict__ b1,
                                                const float* __restrict__ bc,
                                                const float* __restrict__ bb,
                                                float* __restrict__ out) {
  extern __shared__ unsigned short sm[];
  unsigned short* sA  = sm + SA_OFF;    // halo input tile [18 px][32 c]
  unsigned short* sF  = sm + SF_OFF;    // feat tile [16][FSTRIDE] bf16
  unsigned short* sHW = sm + SHW_OFF;   // head weights [64][512] bf16

  const int tid  = threadIdx.x;
  const int wave = tid >> 5;
  const int lane = tid & 31;
  const int mrow = lane & 15;
  const int kb   = (lane >> 4) * 8;     // A/B fragment K sub-base
  const int mhi  = (lane >> 4) * 8;     // C/D fragment M sub-base

  int t = blockIdx.x;
  const int wt = t % WTILES; t /= WTILES;
  const int h  = t % H_;
  const int b  = t / H_;
  const int w0 = wt * TILE_M;

  // ---- TDM: DMA head weights (64KB) global -> LDS, overlapped with conv ----
#if __has_builtin(__builtin_amdgcn_tensor_load_to_lds)
  if (wave == 0) {
    const unsigned lds = (unsigned)(unsigned long long)(uintptr_t)sHW;
    const unsigned long long ga = (unsigned long long)(uintptr_t)hw2;
    u32x4 g0; i32x8 g1; i32x4 g2 = {0,0,0,0}, g3 = {0,0,0,0};
    g0[0] = 1u;                                              // count=1
    g0[1] = lds;                                             // lds_addr
    g0[2] = (unsigned)(ga & 0xFFFFFFFFull);                  // global_addr lo
    g0[3] = (unsigned)((ga >> 32) & 0x1FFFFFFull) | (2u << 30); // hi + type=2
    g1[0] = (3 << 16);                    // data_size = 8B units, no multicast
    g1[1] = (int)(8192u << 16);           // tensor_dim0[15:0] = 8192
    g1[2] = (int)(1u << 16);              // tensor_dim0 hi=0, tensor_dim1=1
    g1[3] = (int)(8192u << 16);           // tile_dim0 = 8192 (x 8B = 64KB)
    g1[4] = 0;                            // tile_dim1/2 unused
    g1[5] = 8192;                         // tensor_dim0_stride
    g1[6] = 0; g1[7] = 0;
#if defined(__clang_major__) && (__clang_major__ >= 23)
    i32x8 g1b = {0,0,0,0,0,0,0,0};
    __builtin_amdgcn_tensor_load_to_lds(g0, g1, g2, g3, g1b, 0);
#else
    __builtin_amdgcn_tensor_load_to_lds(g0, g1, g2, g3, 0);
#endif
  }
#endif

  // hint: this wave's weight slice is hot for the whole K loop (L2 resident)
  __builtin_prefetch(wt2 + (size_t)(wave * 64) * C_, 0, 1);

  // ---- conv: 9 shifted GEMMs, K = 256 each, bf16 WMMA, f32 accum ----
  f32x8 acc[4] = {{0,0,0,0,0,0,0,0},{0,0,0,0,0,0,0,0},
                  {0,0,0,0,0,0,0,0},{0,0,0,0,0,0,0,0}};

  for (int ky = 0; ky < 3; ++ky) {
    const int hy = h + ky - 1;
    const bool rowok = (hy >= 0) && (hy < H_);
    for (int cc = 0; cc < 8; ++cc) {
      __syncthreads();
      if (tid < 144) {                       // stage 18px x 32c halo row
        const int px = tid >> 3, part = tid & 7;
        const int w = w0 - 1 + px;
        uint2 v = make_uint2(0u, 0u);
        if (rowok && (unsigned)w < (unsigned)W_) {
          const unsigned short* src =
              x2 + ((((size_t)b * H_ + hy) * W_ + w) * C_ + cc * 32 + part * 4);
          v = *(const uint2*)src;
        }
        *(uint2*)(sA + px * 32 + part * 4) = v;
      }
      __syncthreads();
#pragma unroll
      for (int kx = 0; kx < 3; ++kx) {
        const int kk = ky * 3 + kx;
        const bf16x16 a = frag16(sA + (mrow + kx) * 32, kb);
        const unsigned short* wrow =
            wt2 + ((size_t)kk * CO_ + mrow) * C_ + cc * 32;
#pragma unroll
        for (int nj = 0; nj < 4; ++nj) {
          const unsigned short* wp = wrow + (size_t)(wave * 64 + nj * 16) * C_;
          const bf16x16 bfr = frag16(wp, kb);
          acc[nj] = __builtin_amdgcn_wmma_f32_16x16x32_bf16(
              false, a, false, bfr, (short)0, acc[nj], false, false);
        }
      }
    }
  }

  // ---- epilogue: bias + ReLU -> bf16 feat tile in LDS ----
  float bias[4];
#pragma unroll
  for (int nj = 0; nj < 4; ++nj) bias[nj] = b1[wave * 64 + nj * 16 + mrow];
#pragma unroll
  for (int nj = 0; nj < 4; ++nj) {
    const int ncol = wave * 64 + nj * 16 + mrow;
#pragma unroll
    for (int r = 0; r < 8; ++r) {
      float v = acc[nj][r] + bias[nj];
      v = v > 0.f ? v : 0.f;
      sF[(mhi + r) * FSTRIDE + ncol] = f32_to_bf16(v);
    }
  }

#if __has_builtin(__builtin_amdgcn_tensor_load_to_lds)
  if (wave == 0) __builtin_amdgcn_s_wait_tensorcnt(0);
#else
  for (int i = tid; i < HEADN * CO_ / 2; i += 256)      // fallback copy
    ((unsigned int*)sHW)[i] = ((const unsigned int*)hw2)[i];
#endif
  __syncthreads();

  // ---- heads: [16 x 512] x [512 x 64] via 16 WMMAs per wave (waves 0-3) ----
  if (wave < 4) {
    f32x8 hacc = {0,0,0,0,0,0,0,0};
    const int o = wave * 16 + mrow;
    const unsigned short* hwr = sHW + (size_t)o * CO_;
    const unsigned short* fr  = sF + (size_t)mrow * FSTRIDE;
#pragma unroll
    for (int k2 = 0; k2 < 16; ++k2) {
      const bf16x16 a  = frag16(fr + k2 * 32, kb);
      const bf16x16 bw = frag16(hwr + k2 * 32, kb);
      hacc = __builtin_amdgcn_wmma_f32_16x16x32_bf16(
          false, a, false, bw, (short)0, hacc, false, false);
    }
    float hb = 0.f;
    if (o < 18) hb = bc[o]; else if (o < 54) hb = bb[o - 18];
    const size_t prow = ((size_t)b * H_ + h) * W_ + w0;
#pragma unroll
    for (int r = 0; r < 8; ++r) {
      const size_t p = prow + (mhi + r);
      const float v = hacc[r] + hb;
      if (o < 18)      out[p * 18 + o] = v;
      else if (o < 54) out[CLS_TOTAL + p * 36 + (o - 18)] = v;
    }
  }
}

// ---------------- host launcher ----------------
extern "C" void kernel_launch(void* const* d_in, const int* in_sizes, int n_in,
                              void* d_out, int out_size, void* d_ws, size_t ws_size,
                              hipStream_t stream) {
  const float* x  = (const float*)d_in[0];
  const float* W1 = (const float*)d_in[1];
  const float* b1 = (const float*)d_in[2];
  const float* Wc = (const float*)d_in[3];
  const float* bc = (const float*)d_in[4];
  const float* Wb = (const float*)d_in[5];
  const float* bb = (const float*)d_in[6];
  float* out = (float*)d_out;
  char* ws = (char*)d_ws;
  unsigned short* x2  = (unsigned short*)(ws);
  unsigned short* wt2 = (unsigned short*)(ws + WT2_OFF);
  unsigned short* hw2 = (unsigned short*)(ws + HW2_OFF);

  dim3 tb(256);
  k_nchw_to_nhwc<<<dim3((W_ + 31) / 32, H_, B_ * 8), tb, 0, stream>>>(x, x2);
  k_w1_repack<<<dim3((9 << 17) / 256), tb, 0, stream>>>(W1, wt2);
  k_head_repack<<<dim3(HEADN * CO_ / 256), tb, 0, stream>>>(Wc, Wb, hw2);

  const size_t smem = (size_t)SMEM_USHORTS * sizeof(unsigned short); // ~83.6 KB
  rpn_main<<<dim3(B_ * H_ * WTILES), tb, smem, stream>>>(x2, wt2, hw2,
                                                         b1, bc, bb, out);
}
